// GraphAttentionLayer_24017457119791
// MI455X (gfx1250) — matrix operand order repaired and checked by reading, not compiled, not run on hardware
//
#include <hip/hip_runtime.h>
#include <math.h>

// GAT layer for MI455X (gfx1250, wave32, WMMA).
// B=8, N=2048, F_in=128, F_out=64.
// Memory-bound on adj (2x134MB ~= 12us at 23.3 TB/s). Dominant GEMM
// (attn @ h, 4.3 GFLOP) uses v_wmma_f32_16x16x32_f16 with f32 accumulate,
// split-K over j (x2 waves for HBM latency hiding), software-pipelined adj
// loads, hoisted B-frag loads, factored softmax (P = exp(e) * cS[j]).

typedef _Float16 v16h __attribute__((ext_vector_type(16)));
typedef float    v8f  __attribute__((ext_vector_type(8)));

#define NB    8
#define NN    2048
#define FIN   128
#define FOUT  64
#define SLOPE 0.2f
#define NEGINF -1e20f
#define CH    16          // i-chunks for the column-stats pass
#define KSPL  2           // split-K factor over j in the attention GEMM

struct F8 { float v[8]; };
static __device__ __forceinline__ F8 load8(const float* __restrict__ p) {
  F8 r;
  float4 lo = *(const float4*)p;
  float4 hi = *(const float4*)(p + 4);
  r.v[0] = lo.x; r.v[1] = lo.y; r.v[2] = lo.z; r.v[3] = lo.w;
  r.v[4] = hi.x; r.v[5] = hi.y; r.v[6] = hi.z; r.v[7] = hi.w;
  return r;
}

// ---------------- K1: h = inp @ W (exact f32) -----------------------------
__global__ __launch_bounds__(256) void gat_h_kernel(
    const float* __restrict__ inp, const float* __restrict__ W,
    float* __restrict__ h) {
  int idx = blockIdx.x * 256 + threadIdx.x;     // over B*N*FOUT
  int f   = idx & (FOUT - 1);
  int row = idx >> 6;                            // b*N + n
  const float* ir = inp + (size_t)row * FIN;
  float acc = 0.f;
#pragma unroll 8
  for (int k = 0; k < FIN; ++k) acc = fmaf(ir[k], W[k * FOUT + f], acc);
  h[(size_t)row * FOUT + f] = acc;
}

// ---------------- K2: s1, s2 and f16 transpose of h -----------------------
__global__ __launch_bounds__(256) void gat_s_hT_kernel(
    const float* __restrict__ h, const float* __restrict__ a,
    float* __restrict__ s1, float* __restrict__ s2,
    _Float16* __restrict__ hT) {
  int row = blockIdx.x * 256 + threadIdx.x;      // b*N + n
  int b = row >> 11, n = row & (NN - 1);
  const float* hr = h + (size_t)row * FOUT;
  _Float16* ht = hT + (size_t)b * FOUT * NN + n; // hT[b][d][n]
  float d1 = 0.f, d2 = 0.f;
#pragma unroll
  for (int d = 0; d < FOUT; ++d) {
    float v = hr[d];
    d1 = fmaf(v, a[d], d1);
    d2 = fmaf(v, a[FOUT + d], d2);
    ht[(size_t)d * NN] = (_Float16)v;            // coalesced along n
  }
  s1[row] = d1;
  s2[row] = d2;
}

// ---------------- K3: per-column online softmax stats (chunked over i) ----
__global__ __launch_bounds__(256) void gat_stats_kernel(
    const float* __restrict__ adj, const float* __restrict__ s1,
    const float* __restrict__ s2, float* __restrict__ pm,
    float* __restrict__ ps) {
  int j     = blockIdx.x * 256 + threadIdx.x;    // column
  int chunk = blockIdx.y;                        // i-chunk
  int b     = blockIdx.z;
  const int ilen = NN / CH;
  float s2v = s2[b * NN + j];
  const float* ab  = adj + ((size_t)b * NN + (size_t)chunk * ilen) * NN + j;
  const float* s1b = s1 + b * NN + chunk * ilen;
  float m = -INFINITY, s = 0.f;
  for (int i = 0; i < ilen; ++i) {
    float av = ab[(size_t)i * NN];               // coalesced across j
    float t  = s1b[i] + s2v;
    float e  = fmaxf(t, SLOPE * t);              // leaky_relu, branch-free
    e = (av > 0.f) ? e : NEGINF;                 // matches reference masking
    float mn = fmaxf(m, e);
    s = s * __expf(m - mn) + __expf(e - mn);
    m = mn;
  }
  pm[((size_t)b * CH + chunk) * NN + j] = m;
  ps[((size_t)b * CH + chunk) * NN + j] = s;
}

// ---------------- K4: combine chunk stats -> cS[j] = exp(-M[j])/S[j] ------
__global__ __launch_bounds__(256) void gat_combine_kernel(
    const float* __restrict__ pm, const float* __restrict__ ps,
    float* __restrict__ cS) {
  int t = blockIdx.x * 256 + threadIdx.x;        // b*N + j
  int b = t >> 11, j = t & (NN - 1);
  float m = -INFINITY, s = 0.f;
#pragma unroll
  for (int c = 0; c < CH; ++c) {
    float mp = pm[((size_t)b * CH + c) * NN + j];
    float sp = ps[((size_t)b * CH + c) * NN + j];
    float mn = fmaxf(m, mp);
    s = s * __expf(m - mn) + sp * __expf(mp - mn);
    m = mn;
  }
  cS[t] = __expf(-m) / s;                        // factored softmax scale
}

// ---------------- K5: partial[z] = attn[:, jz] @ h[jz, :] via WMMA --------
// Block = 128 threads = 4 waves; each wave computes a 16(i) x 64(d) strip
// over one j-half (split-K). A-frag (16x32 f16): P[i][j] = sel(adj>0,
// __expf(lrelu(s1[i]+s2[j])) * cS[j], 0), computed on the fly. Contiguous
// k order as in B: lane (r,g) holds K = 16g+0..7 and 16g+8..15 -> two
// contiguous float4-pairs of adj/s2/cS. B-frag: lane (c,g) holds
// hT[d][j0+16g .. +15] -> one contiguous 32B load; hoisted to iteration
// top so L2 latency hides under the exp pipeline. adj (HBM stream) is
// software double-buffered one iteration ahead + prefetched 1KB ahead.
__global__ __launch_bounds__(128) void gat_attn_gemm_kernel(
    const float* __restrict__ adj, const float* __restrict__ s1v,
    const float* __restrict__ s2v, const float* __restrict__ cSv,
    const _Float16* __restrict__ hT, float* __restrict__ pout) {
  int lane = threadIdx.x & 31;
  int wave = threadIdx.x >> 5;                   // 0..3
  int r = lane & 15, g = lane >> 4;
  int b  = blockIdx.y;
  int z  = blockIdx.z;                           // split-K index
  int i0 = blockIdx.x * 64 + wave * 16;
  int row = i0 + r;
  const int jbeg = z * (NN / KSPL);
  const int jend = jbeg + (NN / KSPL);

  float s1r = s1v[b * NN + row];
  const float*    arow = adj + ((size_t)b * NN + row) * NN;
  const float*    s2b  = s2v + b * NN;
  const float*    cb   = cSv + b * NN;
  const _Float16* hTb  = hT + (size_t)b * FOUT * NN;

  v8f acc0 = {}, acc1 = {}, acc2 = {}, acc3 = {};

  // Prologue: adj chunks for first iteration.
  F8 aA = load8(arow + jbeg + 16 * g);           // K = 16g..16g+7
  F8 aB = load8(arow + jbeg + 16 * g + 8);       // K = 16g+8..+7

  for (int j0 = jbeg; j0 < jend; j0 += 32) {
    int jb = j0 + 16 * g;
    // B frags first: L2-resident, latency hidden by exp pipeline below.
    v16h b0 = *(const v16h*)(hTb + (size_t)(0 * 16 + r) * NN + jb);
    v16h b1 = *(const v16h*)(hTb + (size_t)(1 * 16 + r) * NN + jb);
    v16h b2 = *(const v16h*)(hTb + (size_t)(2 * 16 + r) * NN + jb);
    v16h b3 = *(const v16h*)(hTb + (size_t)(3 * 16 + r) * NN + jb);
    // Next iteration's adj chunks (HBM stream, double-buffered).
    int jn = (j0 + 32 < jend) ? (j0 + 32) : j0;
    F8 nA = load8(arow + jn + 16 * g);
    F8 nB = load8(arow + jn + 16 * g + 8);
    if (j0 + 256 + 32 < NN) __builtin_prefetch(arow + j0 + 256 + 8 * g, 0, 1);
    // Per-column params (L2-hot).
    int c1 = j0 + 16 * g, c2 = c1 + 8;
    F8 sA = load8(s2b + c1);
    F8 sB = load8(s2b + c2);
    F8 cA = load8(cb + c1);
    F8 cB = load8(cb + c2);

    v16h af;
#pragma unroll
    for (int k = 0; k < 8; ++k) {
      {
        float t = s1r + sA.v[k];
        float e = fmaxf(t, SLOPE * t);
        float p = __expf(e) * cA.v[k];
        af[k] = (_Float16)((aA.v[k] > 0.f) ? p : 0.f);
      }
      {
        float t = s1r + sB.v[k];
        float e = fmaxf(t, SLOPE * t);
        float p = __expf(e) * cB.v[k];
        af[8 + k] = (_Float16)((aB.v[k] > 0.f) ? p : 0.f);
      }
    }
    acc0 = __builtin_amdgcn_wmma_f32_16x16x32_f16(false, af, false, b0,
                                                  (short)0, acc0, false, false);
    acc1 = __builtin_amdgcn_wmma_f32_16x16x32_f16(false, af, false, b1,
                                                  (short)0, acc1, false, false);
    acc2 = __builtin_amdgcn_wmma_f32_16x16x32_f16(false, af, false, b2,
                                                  (short)0, acc2, false, false);
    acc3 = __builtin_amdgcn_wmma_f32_16x16x32_f16(false, af, false, b3,
                                                  (short)0, acc3, false, false);
    aA = nA;
    aB = nB;
  }

  // C/D layout: VGPR v, lanes 0-15 -> M=v, lanes 16-31 -> M=8+v; N = lane&15.
  float* ob = pout + ((size_t)z * NB + b) * NN * FOUT;
#pragma unroll
  for (int v = 0; v < 8; ++v) {
    int ro = i0 + 8 * g + v;
    ob[(size_t)ro * FOUT + 0 * 16 + r] = acc0[v];
    ob[(size_t)ro * FOUT + 1 * 16 + r] = acc1[v];
    ob[(size_t)ro * FOUT + 2 * 16 + r] = acc2[v];
    ob[(size_t)ro * FOUT + 3 * 16 + r] = acc3[v];
  }
}

// ---------------- K6: out = gelu(sum of split-K partials) -----------------
__global__ __launch_bounds__(256) void gat_epilogue_kernel(
    const float* __restrict__ p0, const float* __restrict__ p1,
    float* __restrict__ out) {
  size_t idx = (size_t)blockIdx.x * 256 + threadIdx.x;
  float x = p0[idx] + p1[idx];
  out[idx] = 0.5f * x * (1.f + erff(x * 0.70710678f));  // exact gelu
}

extern "C" void kernel_launch(void* const* d_in, const int* in_sizes, int n_in,
                              void* d_out, int out_size, void* d_ws,
                              size_t ws_size, hipStream_t stream) {
  (void)in_sizes; (void)n_in; (void)out_size; (void)ws_size;
  const float* inp = (const float*)d_in[0];   // (8,2048,128)
  const float* adj = (const float*)d_in[1];   // (8,2048,2048)
  const float* W   = (const float*)d_in[2];   // (128,64)
  const float* a   = (const float*)d_in[3];   // (128,1)
  float* out = (float*)d_out;                 // (8,2048,64)

  char* ws = (char*)d_ws;
  size_t off = 0;
  auto carve = [&](size_t bytes) -> char* {
    char* p = ws + off;
    off += (bytes + 255) & ~(size_t)255;
    return p;
  };
  float*    h    = (float*)   carve((size_t)NB * NN * FOUT * 4);        // 4 MB
  _Float16* hT   = (_Float16*)carve((size_t)NB * FOUT * NN * 2);        // 2 MB
  float*    s1   = (float*)   carve((size_t)NB * NN * 4);
  float*    s2   = (float*)   carve((size_t)NB * NN * 4);
  float*    pm   = (float*)   carve((size_t)NB * CH * NN * 4);          // 1 MB
  float*    ps   = (float*)   carve((size_t)NB * CH * NN * 4);          // 1 MB
  float*    cS   = (float*)   carve((size_t)NB * NN * 4);
  float*    pout = (float*)   carve((size_t)KSPL * NB * NN * FOUT * 4); // 8 MB

  gat_h_kernel<<<(NB * NN * FOUT) / 256, 256, 0, stream>>>(inp, W, h);
  gat_s_hT_kernel<<<(NB * NN) / 256, 256, 0, stream>>>(h, a, s1, s2, hT);
  gat_stats_kernel<<<dim3(NN / 256, CH, NB), 256, 0, stream>>>(adj, s1, s2, pm, ps);
  gat_combine_kernel<<<(NB * NN) / 256, 256, 0, stream>>>(pm, ps, cS);
  gat_attn_gemm_kernel<<<dim3(NN / 64, NB, KSPL), 128, 0, stream>>>(
      adj, s1, s2, cS, hT, pout);
  gat_epilogue_kernel<<<(NB * NN * FOUT) / 256, 256, 0, stream>>>(
      pout, pout + (size_t)NB * NN * FOUT, out);
}
